// Decoder_8169027797469
// MI455X (gfx1250) — matrix operand order repaired and checked by reading, not compiled, not run on hardware
//
#include <hip/hip_runtime.h>
#include <hip/hip_bf16.h>
#include <math.h>

// Problem constants (match reference)
#define BB 128
#define TT 1024
#define DD 512
#define EE 256
#define VV 50257

typedef __bf16 bf16_t;
typedef bf16_t v16bf __attribute__((ext_vector_type(16)));
typedef float  v8f   __attribute__((ext_vector_type(8)));

__device__ __forceinline__ float sigmoidf_(float x) { return 1.0f / (1.0f + __expf(-x)); }

// ---------------------------------------------------------------------------
// WMMA bf16 GEMM, N compile-time. One wave computes a 16x64 strip (4 16x16
// tiles) so the A fragment is reused 4x and 4 v_wmma issue per K-step.
// Lane mapping used here (contiguous k, not the ISA table order), lane L: mrow/n-col = L&15, kh = L>>4:
//   A (16x32 bf16): a[t] = K(16*kh + t), a[8+t] = K(16*kh + 8 + t)
//   B (32x16 bf16): b[t] = K(16*kh + t) at column n
//   C/D (16x16 f32): acc[j] -> row m0 + j + 8*kh, col n0 + (L&15)
// Out-of-range columns are CLAMPED (loads stay in-bounds, never predicated);
// the garbage columns are simply not stored.
// ---------------------------------------------------------------------------
template <int N>
__device__ __forceinline__ void wmma_accum4(const float* __restrict__ A,
                                            const float* __restrict__ Bm,
                                            int K, int m0, int n0, int mrow, int kh,
                                            v8f acc[4])
{
    const float* arow = A + (size_t)(m0 + mrow) * K;
    const float* bp[4];
#pragma unroll
    for (int s = 0; s < 4; ++s) {
        int n = n0 + s * 16 + mrow;
        if (n > N - 1) n = N - 1;                 // clamp: safe, unconditional loads
        bp[s] = Bm + (size_t)(16 * kh) * N + n;
    }
    for (int k0 = 0; k0 < K; k0 += 32) {
        v16bf a;
#pragma unroll
        for (int t = 0; t < 8; ++t) {
            a[t]     = (bf16_t)arow[k0 + 16 * kh + t];
            a[8 + t] = (bf16_t)arow[k0 + 16 * kh + 8 + t];
        }
#pragma unroll
        for (int s = 0; s < 4; ++s) {
            v16bf b;
#pragma unroll
            for (int t = 0; t < 16; ++t)
                b[t] = (bf16_t)bp[s][(size_t)t * N];  // const offsets: t*N*4 bytes
            acc[s] = __builtin_amdgcn_wmma_f32_16x16x32_bf16(
                false, a, false, b, (short)0, acc[s], false, false);
            bp[s] += (size_t)32 * N;
        }
    }
}

template <int N, int ACT /*0=none,1=tanh*/>
__global__ __launch_bounds__(128)
void wmma_gemm_kernel(const float* __restrict__ A1, const float* __restrict__ B1, int K1,
                      const float* __restrict__ A2, const float* __restrict__ B2, int K2,
                      const float* __restrict__ bias, float* __restrict__ C, int M)
{
    constexpr int NT = (N + 15) / 16;   // 16-col tiles
    constexpr int NG = (NT + 3) / 4;    // 64-col groups
    const int wave = blockIdx.x * (blockDim.x >> 5) + (threadIdx.x >> 5);
    const int lane = threadIdx.x & 31;
    const int mtiles = M >> 4;
    if (wave >= mtiles * NG) return;
    const int mt = wave / NG, ng = wave % NG;
    const int m0 = mt << 4, n0 = ng << 6;
    const int mrow = lane & 15, kh = lane >> 4;

    v8f acc[4] = {};
    wmma_accum4<N>(A1, B1, K1, m0, n0, mrow, kh, acc);
    if (A2) wmma_accum4<N>(A2, B2, K2, m0, n0, mrow, kh, acc);

#pragma unroll
    for (int s = 0; s < 4; ++s) {
        const int n = n0 + s * 16 + mrow;
        if (n < N) {
            const float bv = bias ? bias[n] : 0.0f;
#pragma unroll
            for (int j = 0; j < 8; ++j) {
                const int r = m0 + j + 8 * kh;
                float v = acc[s][j] + bv;
                if (ACT == 1) v = tanhf(v);
                C[(size_t)r * N + n] = v;
            }
        }
    }
}

// ---------------------------------------------------------------------------
// step_in[b, :] = concat(emb_table[x[b]], last_hidden_att[b])
// ---------------------------------------------------------------------------
__global__ void build_stepin_kernel(const int* __restrict__ x,
                                    const float* __restrict__ emb,
                                    const float* __restrict__ lha,
                                    float* __restrict__ step_in)
{
    const int b = blockIdx.y;
    const int i = blockIdx.x * blockDim.x + threadIdx.x;
    if (i >= EE + DD) return;
    float v = (i < EE) ? emb[(size_t)x[b] * EE + i] : lha[(size_t)b * DD + (i - EE)];
    step_in[(size_t)b * (EE + DD) + i] = v;
}

// ---------------------------------------------------------------------------
// LSTM gates: z=(B,4D) laid out [i|f|g|o]
// ---------------------------------------------------------------------------
__global__ void lstm_pointwise_kernel(const float* __restrict__ z,
                                      const float* __restrict__ c_prev,
                                      float* __restrict__ h_out,
                                      float* __restrict__ c_out)
{
    const int idx = blockIdx.x * blockDim.x + threadIdx.x;
    if (idx >= BB * DD) return;
    const int b = idx / DD, d = idx % DD;
    const float* zr = z + (size_t)b * 4 * DD;
    const float i = zr[d];
    const float f = zr[DD + d];
    const float g = zr[2 * DD + d];
    const float o = zr[3 * DD + d];
    const float c = sigmoidf_(f) * c_prev[idx] + sigmoidf_(i) * tanhf(g);
    h_out[idx] = sigmoidf_(o) * tanhf(c);
    c_out[idx] = c;
}

// ---------------------------------------------------------------------------
// scores[b,t] = dot(h2[b,:], enc[b,t,:]) — one wave per (b,t), float4 loads
// ---------------------------------------------------------------------------
__global__ void scores_kernel(const float* __restrict__ h2,
                              const float* __restrict__ enc,
                              float* __restrict__ scores)
{
    const int gw   = (blockIdx.x * blockDim.x + threadIdx.x) >> 5;
    const int lane = threadIdx.x & 31;
    if (gw >= BB * TT) return;
    const int b = gw / TT, t = gw % TT;
    const float4* e = (const float4*)(enc + ((size_t)b * TT + t) * DD);
    const float4* h = (const float4*)(h2 + (size_t)b * DD);
    float s = 0.0f;
#pragma unroll
    for (int i = 0; i < DD / 128; ++i) {           // DD/4 float4s / 32 lanes
        const float4 ev = e[lane + 32 * i];
        const float4 hv = h[lane + 32 * i];
        s += ev.x * hv.x + ev.y * hv.y + ev.z * hv.z + ev.w * hv.w;
    }
#pragma unroll
    for (int off = 16; off; off >>= 1) s += __shfl_xor(s, off, 32);
    if (lane == 0) scores[gw] = s;
}

// ---------------------------------------------------------------------------
// softmax over T per batch row (block per row)
// ---------------------------------------------------------------------------
__global__ void softmax_T_kernel(const float* __restrict__ scores,
                                 float* __restrict__ align)
{
    __shared__ float red[256];
    const int b = blockIdx.x, tid = threadIdx.x;
    const float* row = scores + (size_t)b * TT;
    float mx = -3.4e38f;
    for (int t = tid; t < TT; t += 256) mx = fmaxf(mx, row[t]);
    red[tid] = mx; __syncthreads();
    for (int s = 128; s; s >>= 1) { if (tid < s) red[tid] = fmaxf(red[tid], red[tid + s]); __syncthreads(); }
    mx = red[0]; __syncthreads();
    float sum = 0.0f;
    for (int t = tid; t < TT; t += 256) {
        float e = __expf(row[t] - mx);
        align[(size_t)b * TT + t] = e;
        sum += e;
    }
    red[tid] = sum; __syncthreads();
    for (int s = 128; s; s >>= 1) { if (tid < s) red[tid] += red[tid + s]; __syncthreads(); }
    const float inv = 1.0f / red[0];
    for (int t = tid; t < TT; t += 256) align[(size_t)b * TT + t] *= inv;
}

// ---------------------------------------------------------------------------
// context[b,d] = sum_t align[b,t]*enc[b,t,d]; also builds concat=[context|h2]
// ---------------------------------------------------------------------------
__global__ void context_kernel(const float* __restrict__ align,
                               const float* __restrict__ enc,
                               const float* __restrict__ h2,
                               float* __restrict__ concat)
{
    const int idx = blockIdx.x * blockDim.x + threadIdx.x;
    if (idx >= BB * DD) return;
    const int b = idx / DD, d = idx % DD;
    const float* e = enc + (size_t)b * TT * DD + d;
    const float* a = align + (size_t)b * TT;
    float s = 0.0f;
    for (int t = 0; t < TT; ++t) s += a[t] * e[(size_t)t * DD];
    concat[(size_t)b * 2 * DD + d] = s;
    concat[(size_t)b * 2 * DD + DD + d] = h2[idx];
}

__global__ void zero_kernel(float* __restrict__ p, size_t n)
{
    size_t i = (size_t)blockIdx.x * blockDim.x + threadIdx.x;
    if (i < n) p[i] = 0.0f;
}

__global__ void scatter_kernel(const int* __restrict__ ex,
                               const float* __restrict__ align,
                               float* __restrict__ copyp)
{
    const int i = blockIdx.x * blockDim.x + threadIdx.x;
    if (i >= BB * TT) return;
    const int b = i / TT;
    atomicAdd(&copyp[(size_t)b * VV + ex[i]], align[i]);
}

// switch[b] = sigmoid(dot(concat[b,:2D], fc3_w) + fc3_b) — one wave per row
__global__ void switch_kernel(const float* __restrict__ concat,
                              const float* __restrict__ w,
                              const float* __restrict__ bias,
                              float* __restrict__ sw)
{
    const int b = blockIdx.x, lane = threadIdx.x;
    float s = 0.0f;
    for (int d = lane; d < 2 * DD; d += 32) s += concat[(size_t)b * 2 * DD + d] * w[d];
#pragma unroll
    for (int off = 16; off; off >>= 1) s += __shfl_xor(s, off, 32);
    if (lane == 0) sw[b] = sigmoidf_(s + bias[0]);
}

// per-row max and 1/sum(exp(x-max)) over V (block per row)
__global__ void rowstats_kernel(const float* __restrict__ logits,
                                float* __restrict__ stats)
{
    __shared__ float red[256];
    const int b = blockIdx.x, tid = threadIdx.x;
    const float* row = logits + (size_t)b * VV;
    float mx = -3.4e38f;
    for (int i = tid; i < VV; i += 256) mx = fmaxf(mx, row[i]);
    red[tid] = mx; __syncthreads();
    for (int s = 128; s; s >>= 1) { if (tid < s) red[tid] = fmaxf(red[tid], red[tid + s]); __syncthreads(); }
    mx = red[0]; __syncthreads();
    float sum = 0.0f;
    for (int i = tid; i < VV; i += 256) sum += __expf(row[i] - mx);
    red[tid] = sum; __syncthreads();
    for (int s = 128; s; s >>= 1) { if (tid < s) red[tid] += red[tid + s]; __syncthreads(); }
    if (tid == 0) { stats[2 * b] = mx; stats[2 * b + 1] = 1.0f / red[0]; }
}

// gen (in-place on logits) = softmax; result = gen*(1-sw) + copy*sw
__global__ void finalize_kernel(const float* __restrict__ stats,
                                const float* __restrict__ sw,
                                const float* __restrict__ copyp,
                                float* __restrict__ gen,
                                float* __restrict__ result)
{
    size_t i = (size_t)blockIdx.x * blockDim.x + threadIdx.x;
    if (i >= (size_t)BB * VV) return;
    const int b = (int)(i / VV);
    const float g = __expf(gen[i] - stats[2 * b]) * stats[2 * b + 1];
    gen[i] = g;
    const float s = sw[b];
    result[i] = g * (1.0f - s) + copyp[i] * s;
}

// ---------------------------------------------------------------------------
static inline int gemm_blocks(int M, int N)
{
    const int nt = (N + 15) / 16;
    const int ng = (nt + 3) / 4;
    const int waves = (M / 16) * ng;
    return (waves + 3) / 4;   // 4 waves (128 threads) per block
}

extern "C" void kernel_launch(void* const* d_in, const int* in_sizes, int n_in,
                              void* d_out, int out_size, void* d_ws, size_t ws_size,
                              hipStream_t stream)
{
    (void)in_sizes; (void)n_in; (void)out_size; (void)ws_size;

    const int*   encoder_x = (const int*)  d_in[0];
    const int*   x         = (const int*)  d_in[1];
    const float* h1_0      = (const float*)d_in[2];
    const float* c1_0      = (const float*)d_in[3];
    const float* h2_0      = (const float*)d_in[4];
    const float* c2_0      = (const float*)d_in[5];
    const float* lha       = (const float*)d_in[6];
    const float* enc       = (const float*)d_in[7];
    const float* emb       = (const float*)d_in[8];
    const float* W1k       = (const float*)d_in[9];
    const float* W1r       = (const float*)d_in[10];
    const float* b1        = (const float*)d_in[11];
    const float* W2k       = (const float*)d_in[12];
    const float* W2r       = (const float*)d_in[13];
    const float* b2        = (const float*)d_in[14];
    const float* fc1_w     = (const float*)d_in[15];
    const float* fc1_b     = (const float*)d_in[16];
    const float* fc2_w     = (const float*)d_in[17];
    const float* fc2_b     = (const float*)d_in[18];
    const float* fc3_w     = (const float*)d_in[19];
    const float* fc3_b     = (const float*)d_in[20];

    // d_out layout (floats), tuple order of the reference:
    const size_t BV = (size_t)BB * VV, BD = (size_t)BB * DD, BT = (size_t)BB * TT;
    float* out        = (float*)d_out;
    float* o_result   = out;
    float* o_gen      = out + BV;            // logits computed here, softmax in place
    float* o_switch   = out + 2 * BV;        // (B,1)
    float* o_copy     = out + 2 * BV + BB;
    float* o_hidatt   = out + 3 * BV + BB;
    float* o_align    = o_hidatt + BD;
    float* o_h1       = o_align + BT;
    float* o_c1       = o_h1 + BD;
    float* o_h2       = o_c1 + BD;
    float* o_c2       = o_h2 + BD;

    // workspace layout (floats)
    float* ws        = (float*)d_ws;
    float* w_stepin  = ws;                                   // 128*768
    float* w_z       = w_stepin + (size_t)BB * (EE + DD);    // 128*2048 (reused as scores)
    float* w_concat  = w_z + (size_t)BB * 4 * DD;            // 128*1024
    float* w_stats   = w_concat + (size_t)BB * 2 * DD;       // 256

    // 1) step_in = [emb[x], last_hidden_att]
    build_stepin_kernel<<<dim3((EE + DD + 255) / 256, BB), 256, 0, stream>>>(
        x, emb, lha, w_stepin);

    // 2) z1 = step_in@W1k + h1_0@W1r + b1  (M=128, N=2048, K=768/512)
    wmma_gemm_kernel<4 * DD, 0><<<gemm_blocks(BB, 4 * DD), 128, 0, stream>>>(
        w_stepin, W1k, EE + DD, h1_0, W1r, DD, b1, w_z, BB);
    lstm_pointwise_kernel<<<(BB * DD + 255) / 256, 256, 0, stream>>>(w_z, c1_0, o_h1, o_c1);

    // 3) z2 = h1@W2k + h2_0@W2r + b2
    wmma_gemm_kernel<4 * DD, 0><<<gemm_blocks(BB, 4 * DD), 128, 0, stream>>>(
        o_h1, W2k, DD, h2_0, W2r, DD, b2, w_z, BB);
    lstm_pointwise_kernel<<<(BB * DD + 255) / 256, 256, 0, stream>>>(w_z, c2_0, o_h2, o_c2);

    // 4) attention scores + softmax -> alignment
    scores_kernel<<<(BB * TT * 32) / 256, 256, 0, stream>>>(o_h2, enc, w_z /*scores*/);
    softmax_T_kernel<<<BB, 256, 0, stream>>>(w_z, o_align);

    // 5) context + concat[context|h2]
    context_kernel<<<(BB * DD + 255) / 256, 256, 0, stream>>>(o_align, enc, o_h2, w_concat);

    // 6) copy_prob: zero then scatter-add alignment at encoder_x
    zero_kernel<<<(unsigned)((BV + 255) / 256), 256, 0, stream>>>(o_copy, BV);
    scatter_kernel<<<(BB * TT + 255) / 256, 256, 0, stream>>>(encoder_x, o_align, o_copy);

    // 7) hidden_att = tanh(concat@fc1_w + fc1_b)   (M=128, N=512, K=1024)
    wmma_gemm_kernel<DD, 1><<<gemm_blocks(BB, DD), 128, 0, stream>>>(
        w_concat, fc1_w, 2 * DD, nullptr, nullptr, 0, fc1_b, o_hidatt, BB);

    // 8) switch_prob = sigmoid(concat@fc3_w + fc3_b)
    switch_kernel<<<BB, 32, 0, stream>>>(w_concat, fc3_w, fc3_b, o_switch);

    // 9) logits = hidden_att@fc2_w + fc2_b  (M=128, N=50257, K=512) into gen region
    wmma_gemm_kernel<VV, 0><<<gemm_blocks(BB, VV), 128, 0, stream>>>(
        o_hidatt, fc2_w, DD, nullptr, nullptr, 0, fc2_b, o_gen, BB);

    // 10) softmax over V (stats pass) then finalize: gen + result
    rowstats_kernel<<<BB, 256, 0, stream>>>(o_gen, w_stats);
    finalize_kernel<<<(unsigned)((BV + 255) / 256), 256, 0, stream>>>(
        w_stats, o_switch, o_copy, o_gen, o_result);
}
